// MambaEncoder_8254927143300
// MI455X (gfx1250) — matrix operand order repaired and checked by reading, not compiled, not run on hardware
//
#include <hip/hip_runtime.h>
#include <math.h>

#define D_MODEL  256
#define D_INNER  512
#define D_STATE  32
#define D_CONV   4
#define NHEADS   8
#define HEADDIM  64
#define NLAYERS  2
#define CONV_DIM 576          // D_INNER + 2*D_STATE
#define ZXDIM    1096         // 2*D_INNER + 2*D_STATE + NHEADS
#define BSZ      4
#define SEQ      2048
#define ROWS     (BSZ * SEQ)  // 8192
#define EPSF     1e-5f

typedef float    v2f  __attribute__((ext_vector_type(2)));
typedef float    v8f  __attribute__((ext_vector_type(8)));
typedef _Float16 v16h __attribute__((ext_vector_type(16)));

#if defined(__HIP_DEVICE_COMPILE__) && __has_builtin(__builtin_amdgcn_wmma_f32_16x16x4_f32)
#define WMMA_MODE 1   // native fp32 matrix op: V_WMMA_F32_16X16X4_F32
#elif defined(__HIP_DEVICE_COMPILE__) && __has_builtin(__builtin_amdgcn_wmma_f32_16x16x32_f16)
#define WMMA_MODE 2   // f16 inputs, fp32 accumulate
#else
#define WMMA_MODE 0   // host parse / no builtin
#endif

// ---------------------------------------------------------------- reductions
__device__ __forceinline__ float blockReduceSum256(float v, float* sm) {
  #pragma unroll
  for (int o = 16; o > 0; o >>= 1) v += __shfl_xor(v, o, 32);
  int wid = threadIdx.x >> 5, lane = threadIdx.x & 31;
  if (lane == 0) sm[wid] = v;
  __syncthreads();
  float t = (threadIdx.x < 8) ? sm[threadIdx.x] : 0.f;
  if (wid == 0) {
    #pragma unroll
    for (int o = 16; o > 0; o >>= 1) t += __shfl_xor(t, o, 32);
    if (lane == 0) sm[8] = t;
  }
  __syncthreads();
  float r = sm[8];
  __syncthreads();
  return r;
}

// ---------------------------------------------------------------- LayerNorm
// one block (256 threads) per token row; mask!=null => final masked LN
__global__ void ln_kernel(const float* __restrict__ in, float* __restrict__ out,
                          const float* __restrict__ w, const float* __restrict__ b,
                          const int* __restrict__ mask) {
  __shared__ float sm[9];
  int row = blockIdx.x, tid = threadIdx.x;
  float v  = in[(size_t)row * D_MODEL + tid];
  float mu = blockReduceSum256(v, sm) * (1.f / D_MODEL);
  float d  = v - mu;
  float var = blockReduceSum256(d * d, sm) * (1.f / D_MODEL);
  float r  = d * rsqrtf(var + EPSF) * w[tid] + b[tid];
  if (mask && mask[row]) r = 0.f;
  out[(size_t)row * D_MODEL + tid] = r;
}

// ---------------------------------------------------------------- WMMA GEMM
// C[m][n] = sum_k A[m][k] * W[n][k]   (row-major, K inner for both)
// grid: (M/16, ceil(N/32), 2 dirs); block: 32 (one wave, one 16x32 C tile:
// one A fragment feeds two B fragments / two accumulators).
// Out-of-range B columns load a clamped valid row (branchless inner loop);
// their garbage C columns are simply never stored.
// flipDir1: for dir==1 read A at time-reversed row (bidirectional in_proj)
__global__ void __launch_bounds__(32)
gemm_wmma(const float* __restrict__ A, long long Astride,
          const float* __restrict__ W, long long Wstride,
          float* __restrict__ C, long long Cstride,
          int M, int N, int K, int flipDir1) {
  const int lane = threadIdx.x;
  const int mloc = lane & 15;     // row within A-frag / col within B-frag
  const int hh   = lane >> 4;     // lane-half selects K sub-block
  const int dir  = blockIdx.z;
  const float* Ad = A + (size_t)dir * Astride;
  const float* Wd = W + (size_t)dir * Wstride;
  float*       Cd = C + (size_t)dir * Cstride;
  const bool flip = (flipDir1 != 0) && (dir == 1);

  int m = blockIdx.x * 16 + mloc;
  int am = m;
  if (flip) { int bb_ = m / SEQ, l = m % SEQ; am = bb_ * SEQ + (SEQ - 1 - l); }
  const float* Arow = Ad + (size_t)am * K;

  const int n0 = blockIdx.y * 32 + mloc;        // first B column
  const int n1 = n0 + 16;                       // second B column
  const int n0c = (n0 < N) ? n0 : (N - 1);      // clamp: branchless loads
  const int n1c = (n1 < N) ? n1 : (N - 1);
  const float* Wrow0 = Wd + (size_t)n0c * K;
  const float* Wrow1 = Wd + (size_t)n1c * K;

  v8f acc0 = {0.f, 0.f, 0.f, 0.f, 0.f, 0.f, 0.f, 0.f};
  v8f acc1 = {0.f, 0.f, 0.f, 0.f, 0.f, 0.f, 0.f, 0.f};

#if WMMA_MODE == 1
  // A 16x4 fp32: lane holds A[m][k0+2*hh+{0,1}]; B 4x16: W[n][k0+2*hh+{0,1}]
  for (int k0 = 0; k0 < K; k0 += 4) {
    if ((k0 & 63) == 0) __builtin_prefetch(Arow + k0 + 64, 0, 0);
    int ka = k0 + 2 * hh;
    v2f a  = *(const v2f*)(Arow  + ka);
    v2f b0 = *(const v2f*)(Wrow0 + ka);
    v2f b1 = *(const v2f*)(Wrow1 + ka);
    acc0 = __builtin_amdgcn_wmma_f32_16x16x4_f32(false, a, false, b0,
                                                 (short)0, acc0, false, false);
    acc1 = __builtin_amdgcn_wmma_f32_16x16x4_f32(false, a, false, b1,
                                                 (short)0, acc1, false, false);
  }
#elif WMMA_MODE == 2
  for (int k0 = 0; k0 < K; k0 += 32) {
    v16h a, b0, b1;
    #pragma unroll
    for (int i = 0; i < 8; i++) {
      a[i]     = (_Float16)Arow[k0 + 16 * hh + i];
      a[i + 8] = (_Float16)Arow[k0 + 16 * hh + 8 + i];
    }
    #pragma unroll
    for (int i = 0; i < 16; i++) {
      b0[i] = (_Float16)Wrow0[k0 + 16 * hh + i];
      b1[i] = (_Float16)Wrow1[k0 + 16 * hh + i];
    }
    acc0 = __builtin_amdgcn_wmma_f32_16x16x32_f16(false, a, false, b0,
                                                  (short)0, acc0, false, false);
    acc1 = __builtin_amdgcn_wmma_f32_16x16x32_f16(false, a, false, b1,
                                                  (short)0, acc1, false, false);
  }
#else
  for (int r = 0; r < 8; r++) {
    int mm = blockIdx.x * 16 + r + 8 * hh;
    int amm = mm;
    if (flip) { int bb_ = mm / SEQ, l = mm % SEQ; amm = bb_ * SEQ + (SEQ - 1 - l); }
    const float* Ar = Ad + (size_t)amm * K;
    float s0 = 0.f, s1 = 0.f;
    for (int k = 0; k < K; k++) { s0 += Ar[k] * Wrow0[k]; s1 += Ar[k] * Wrow1[k]; }
    acc0[r] = s0; acc1[r] = s1;
  }
#endif

  // C/D layout: VGPR r -> M = r + 8*hh, N = lane&15
  if (n0 < N) {
    #pragma unroll
    for (int r = 0; r < 8; r++) {
      int mm = blockIdx.x * 16 + r + 8 * hh;
      Cd[(size_t)mm * N + n0] = acc0[r];
    }
  }
  if (n1 < N) {
    #pragma unroll
    for (int r = 0; r < 8; r++) {
      int mm = blockIdx.x * 16 + r + 8 * hh;
      Cd[(size_t)mm * N + n1] = acc1[r];
    }
  }
}

// ---------------------------------------------------------------- conv + silu
__global__ void conv_kernel(const float* __restrict__ zx, float* __restrict__ xc,
                            const float* __restrict__ cw, const float* __restrict__ cb) {
  int idx = blockIdx.x * blockDim.x + threadIdx.x;
  if (idx >= 2 * ROWS * CONV_DIM) return;
  int c   = idx % CONV_DIM;
  int rem = idx / CONV_DIM;
  int row = rem % ROWS;
  int dir = rem / ROWS;
  int l   = row % SEQ;
  const float* zb = zx + (size_t)dir * ROWS * ZXDIM;
  const float* wr = cw + ((size_t)dir * CONV_DIM + c) * D_CONV;
  float acc = cb[dir * CONV_DIM + c];
  #pragma unroll
  for (int i = 0; i < D_CONV; i++) {
    int tl = l - (D_CONV - 1) + i;
    if (tl >= 0)
      acc += zb[(size_t)(row - (D_CONV - 1) + i) * ZXDIM + D_INNER + c] * wr[i];
  }
  acc = acc / (1.f + __expf(-acc));            // silu
  xc[idx] = acc;
}

// ---------------------------------------------------------------- dt = softplus
__global__ void dt_kernel(const float* __restrict__ zx, float* __restrict__ dtb,
                          const float* __restrict__ dt_bias_l) {
  int idx = blockIdx.x * blockDim.x + threadIdx.x;
  if (idx >= 2 * ROWS * NHEADS) return;
  int h   = idx % NHEADS;
  int rem = idx / NHEADS;
  int row = rem % ROWS;
  int dir = rem / ROWS;
  float v = zx[((size_t)dir * ROWS + row) * ZXDIM + D_INNER + CONV_DIM + h]
          + dt_bias_l[dir * NHEADS + h];
  dtb[idx] = (v > 20.f) ? v : log1pf(__expf(v));
}

// ---------------------------------------------------------------- selective scan
// one block per (head, batch, dir); thread owns 8 state cols of one state row
__global__ void scan_kernel(const float* __restrict__ xc, const float* __restrict__ dtb,
                            const float* __restrict__ A_log_l, float* __restrict__ yb) {
  int head = blockIdx.x, b = blockIdx.y, dir = blockIdx.z;
  const float* xcd = xc  + (size_t)dir * ROWS * CONV_DIM;
  const float* dtd = dtb + (size_t)dir * ROWS * NHEADS;
  float*       ybd = yb  + (size_t)dir * ROWS * D_INNER;
  float Aneg = -__expf(A_log_l[dir * NHEADS + head]);

  int tid = threadIdx.x;
  int p = tid >> 2, c = tid & 3, n0 = c * 8;
  __shared__ float sB[D_STATE], sC[D_STATE], sdt;
  float h[8];
  #pragma unroll
  for (int j = 0; j < 8; j++) h[j] = 0.f;

  for (int t = 0; t < SEQ; t++) {
    size_t row = (size_t)b * SEQ + t;
    const float* xrow = xcd + row * CONV_DIM;
    __syncthreads();
    if (tid < D_STATE)            sB[tid]            = xrow[D_INNER + tid];
    else if (tid < 2 * D_STATE)   sC[tid - D_STATE]  = xrow[D_INNER + tid];
    else if (tid == 2 * D_STATE)  sdt                = dtd[row * NHEADS + head];
    __syncthreads();
    float dtv  = sdt;
    float dAv  = __expf(dtv * Aneg);
    float coef = dtv * xrow[head * HEADDIM + p];
    float ysum = 0.f;
    #pragma unroll
    for (int j = 0; j < 8; j++) {
      h[j] = h[j] * dAv + coef * sB[n0 + j];
      ysum += h[j] * sC[n0 + j];
    }
    ysum += __shfl_xor(ysum, 1, 32);
    ysum += __shfl_xor(ysum, 2, 32);
    if (c == 0) ybd[row * D_INNER + head * HEADDIM + p] = ysum;
  }
}

// ---------------------------------------------------------------- gate + RMSNorm
__global__ void gate_kernel(const float* __restrict__ zx, const float* __restrict__ xc,
                            float* __restrict__ yb, const float* __restrict__ Dv_l,
                            const float* __restrict__ gw_l) {
  __shared__ float sm[9];
  int row = blockIdx.x, dir = blockIdx.y, tid = threadIdx.x;
  const float* zrow = zx + ((size_t)dir * ROWS + row) * ZXDIM;
  const float* xrow = xc + ((size_t)dir * ROWS + row) * CONV_DIM;
  float*       yrow = yb + ((size_t)dir * ROWS + row) * D_INNER;
  int e0 = tid, e1 = tid + 256;
  float y0, y1, ss = 0.f;
  {
    float ys = yrow[e0] + Dv_l[dir * NHEADS + (e0 >> 6)] * xrow[e0];
    float z  = zrow[e0];
    y0 = ys * (z / (1.f + __expf(-z)));
    ss += y0 * y0;
  }
  {
    float ys = yrow[e1] + Dv_l[dir * NHEADS + (e1 >> 6)] * xrow[e1];
    float z  = zrow[e1];
    y1 = ys * (z / (1.f + __expf(-z)));
    ss += y1 * y1;
  }
  float tot = blockReduceSum256(ss, sm);
  float r = rsqrtf(tot * (1.f / D_INNER) + EPSF);
  yrow[e0] = y0 * r * gw_l[dir * D_INNER + e0];
  yrow[e1] = y1 * r * gw_l[dir * D_INNER + e1];
}

// ---------------------------------------------------------------- residual
__global__ void residual_kernel(float* __restrict__ xres, const float* __restrict__ po,
                                const int* __restrict__ mask) {
  int idx = blockIdx.x * blockDim.x + threadIdx.x;
  if (idx >= ROWS * D_MODEL) return;
  int d = idx % D_MODEL, row = idx / D_MODEL;
  if (mask[row]) return;
  int b = row / SEQ, l = row % SEQ;
  int rr = b * SEQ + (SEQ - 1 - l);                 // un-flip reverse stream
  xres[idx] += po[idx] + po[(size_t)ROWS * D_MODEL + (size_t)rr * D_MODEL + d];
}

// ---------------------------------------------------------------- launch
extern "C" void kernel_launch(void* const* d_in, const int* in_sizes, int n_in,
                              void* d_out, int out_size, void* d_ws, size_t ws_size,
                              hipStream_t stream) {
  (void)in_sizes; (void)n_in; (void)out_size; (void)ws_size;
  const float* x       = (const float*)d_in[0];
  const int*   mask    = (const int*)  d_in[1];
  const float* in_w    = (const float*)d_in[2];
  const float* conv_w  = (const float*)d_in[3];
  const float* conv_b  = (const float*)d_in[4];
  const float* dt_bias = (const float*)d_in[5];
  const float* A_log   = (const float*)d_in[6];
  const float* Dvec    = (const float*)d_in[7];
  const float* gnorm   = (const float*)d_in[8];
  const float* out_w   = (const float*)d_in[9];
  const float* ln_w    = (const float*)d_in[10];
  const float* ln_b    = (const float*)d_in[11];
  float* out = (float*)d_out;

  float* xres = (float*)d_ws;
  float* hbuf = xres + (size_t)ROWS * D_MODEL;
  float* zx   = hbuf + (size_t)ROWS * D_MODEL;
  float* xc   = zx   + (size_t)2 * ROWS * ZXDIM;
  float* dtb  = xc   + (size_t)2 * ROWS * CONV_DIM;
  float* yb   = dtb  + (size_t)2 * ROWS * NHEADS;
  float* po   = yb   + (size_t)2 * ROWS * D_INNER;

  hipMemcpyAsync(xres, x, (size_t)ROWS * D_MODEL * sizeof(float),
                 hipMemcpyDeviceToDevice, stream);

  for (int layer = 0; layer < NLAYERS; layer++) {
    const float* in_w_l   = in_w    + (size_t)layer * 2 * ZXDIM * D_MODEL;
    const float* conv_w_l = conv_w  + (size_t)layer * 2 * CONV_DIM * D_CONV;
    const float* conv_b_l = conv_b  + (size_t)layer * 2 * CONV_DIM;
    const float* dt_b_l   = dt_bias + (size_t)layer * 2 * NHEADS;
    const float* A_log_l  = A_log   + (size_t)layer * 2 * NHEADS;
    const float* Dvec_l   = Dvec    + (size_t)layer * 2 * NHEADS;
    const float* gnorm_l  = gnorm   + (size_t)layer * 2 * D_INNER;
    const float* out_w_l  = out_w   + (size_t)layer * 2 * D_MODEL * D_INNER;

    ln_kernel<<<ROWS, D_MODEL, 0, stream>>>(xres, hbuf, ln_w, ln_b, (const int*)nullptr);

    // in_proj: M=8192, N=1096, K=256; dir1 reads time-reversed LN rows
    gemm_wmma<<<dim3(ROWS / 16, (ZXDIM + 31) / 32, 2), 32, 0, stream>>>(
        hbuf, 0LL, in_w_l, (long long)ZXDIM * D_MODEL,
        zx, (long long)ROWS * ZXDIM, ROWS, ZXDIM, D_MODEL, 1);

    conv_kernel<<<(2 * ROWS * CONV_DIM + 255) / 256, 256, 0, stream>>>(zx, xc, conv_w_l, conv_b_l);
    dt_kernel<<<(2 * ROWS * NHEADS + 255) / 256, 256, 0, stream>>>(zx, dtb, dt_b_l);
    scan_kernel<<<dim3(NHEADS, BSZ, 2), 256, 0, stream>>>(xc, dtb, A_log_l, yb);
    gate_kernel<<<dim3(ROWS, 2), 256, 0, stream>>>(zx, xc, yb, Dvec_l, gnorm_l);

    // out_proj: M=8192, N=256, K=512
    gemm_wmma<<<dim3(ROWS / 16, D_MODEL / 32, 2), 32, 0, stream>>>(
        yb, (long long)ROWS * D_INNER, out_w_l, (long long)D_MODEL * D_INNER,
        po, (long long)ROWS * D_MODEL, ROWS, D_MODEL, D_INNER, 0);

    residual_kernel<<<(ROWS * D_MODEL + 255) / 256, 256, 0, stream>>>(xres, po, mask);
  }

  ln_kernel<<<ROWS, D_MODEL, 0, stream>>>(xres, out, ln_w, ln_b, mask);
}
